// MogGRU_12790412607596
// MI455X (gfx1250) — matrix operand order repaired and checked by reading, not compiled, not run on hardware
//
#include <hip/hip_runtime.h>
#include <hip/hip_bf16.h>

// ---------------------------------------------------------------------------
// Bidirectional Mogrifier GRU for MI455X (gfx1250, wave32, WMMA).
// B=16, T=1024, D=H=768.
//
// Strategy:
//  * Serial scan => only 2-way (fw/bw) parallelism. One persistent 512-thread
//    workgroup per direction; state (h, xm, hm, xt) lives in LDS (f32), gate
//    buffer gh in LDS (f16). Total dynamic LDS = 264KB (<320KB/WGP).
//  * All GEMMs use v_wmma_f32_16x16x32_f16; M=16 == batch, one tile row.
//  * Weights are packed ONCE per launch into the exact WMMA B-fragment
//    lane layout as f16 => hot-loop B loads are coalesced b128 from L2
//    (weights ~19MB packed, fully L2 resident on the 192MB L2).
//  * gi GEMM keeps the r/z/n chunks for a wave's columns in registers and
//    fuses the GRU gate math + masked state update + output store.
//  * Elementwise sigmoid/tanh use v_exp_f32 + v_rcp_f32 (no precise-div
//    chains) to keep the serial inter-WMMA epilogues short.
// ---------------------------------------------------------------------------

typedef __attribute__((ext_vector_type(16))) _Float16 v16h;
typedef __attribute__((ext_vector_type(8)))  float    v8f;
typedef __attribute__((ext_vector_type(4)))  float    v4f;

#define BB  16
#define TT  1024
#define DD  768
#define HH  768
#define KT  24          // 768 / 32 k-tiles
#define NTHREADS 512    // 16 waves

__device__ __forceinline__ v8f wmma16(v16h a, v16h b, v8f c) {
  return __builtin_amdgcn_wmma_f32_16x16x32_f16(
      false, a, false, b, (short)0, c, false, false);
}

// Fast transcendentals: v_exp_f32 + v_rcp_f32 (avoid IEEE div chains).
__device__ __forceinline__ float sigf(float u) {
  return __builtin_amdgcn_rcpf(1.f + __expf(-u));
}
__device__ __forceinline__ float tanh_fast(float u) {
  return 2.f * __builtin_amdgcn_rcpf(1.f + __expf(-2.f * u)) - 1.f;
}

// A fragment (16x32 f16) gathered from an LDS f32 activation array [16][768].
// K order used here and in pack_w_kernel: lane(m = lane&15, g = lane>>4):
//   halves 0..15 -> K = k0 + g*16 + 0..15    (contiguous!)
__device__ __forceinline__ v16h load_a(const float* __restrict__ act,
                                       int k0, int lane) {
  const int m = lane & 15, g = lane >> 4;
  const v4f* p = (const v4f*)(act + m * 768 + k0 + g * 16);
  v4f x0 = p[0], x1 = p[1], y0 = p[2], y1 = p[3];
  v16h a;
  a[0]=(_Float16)x0[0]; a[1]=(_Float16)x0[1]; a[2]=(_Float16)x0[2]; a[3]=(_Float16)x0[3];
  a[4]=(_Float16)x1[0]; a[5]=(_Float16)x1[1]; a[6]=(_Float16)x1[2]; a[7]=(_Float16)x1[3];
  a[8]=(_Float16)y0[0]; a[9]=(_Float16)y0[1]; a[10]=(_Float16)y0[2]; a[11]=(_Float16)y0[3];
  a[12]=(_Float16)y1[0]; a[13]=(_Float16)y1[1]; a[14]=(_Float16)y1[2]; a[15]=(_Float16)y1[3];
  return a;
}

// B fragment: pre-packed so each lane's 16 halves are contiguous 32B.
__device__ __forceinline__ v16h load_b(const _Float16* __restrict__ p,
                                       int frag, int lane) {
  return *(const v16h*)(p + ((size_t)frag << 9) + (lane << 4));
}

// One mogrifier GEMM: dst[m][n] = 2*sigmoid((A @ Bp)[m][n]) * mul[m][n]
// A: LDS f32 [16][768]; Bp: packed f16 (768x768); 48 N-tiles, 3 per wave.
__device__ __forceinline__ void mog_gemm(const float* __restrict__ A_s,
                                         const _Float16* __restrict__ Bp,
                                         const float* __restrict__ mul_s,
                                         float* __restrict__ dst_s,
                                         int wave, int lane) {
  __syncthreads();  // A_s / mul_s produced by previous phase
  v8f acc[3] = {};
  const int nt0 = wave * 3;
  for (int kt = 0; kt < KT; ++kt) {
    v16h a = load_a(A_s, kt * 32, lane);
#pragma unroll
    for (int i = 0; i < 3; ++i) {
      v16h b = load_b(Bp, (nt0 + i) * KT + kt, lane);
      acc[i] = wmma16(a, b, acc[i]);
    }
  }
  const int g = lane >> 4, nloc = lane & 15;
#pragma unroll
  for (int i = 0; i < 3; ++i) {
    const int n = (nt0 + i) * 16 + nloc;
#pragma unroll
    for (int v = 0; v < 8; ++v) {
      const int m = v + 8 * g;
      float s = 2.f * sigf(acc[i][v]);
      dst_s[m * 768 + n] = s * mul_s[m * 768 + n];  // element owned by this lane
    }
  }
}

__global__ void __launch_bounds__(NTHREADS, 1)
mog_gru_kernel(const float* __restrict__ xg, const int* __restrict__ lengths,
               const _Float16* __restrict__ wbase,
               const float* __restrict__ bih_fw, const float* __restrict__ bhh_fw,
               const float* __restrict__ bih_bw, const float* __restrict__ bhh_bw,
               float* __restrict__ out_fw, float* __restrict__ out_bw) {
  extern __shared__ char smem[];
  float*    h_s  = (float*)smem;            // 16*768 f32  (48KB)
  float*    xm_s = h_s  + 16 * 768;         // 48KB
  float*    hm_s = xm_s + 16 * 768;         // 48KB
  float*    xt_s = hm_s + 16 * 768;         // 48KB
  _Float16* gh_s = (_Float16*)(xt_s + 16 * 768);  // 16*2304 f16 (72KB)
  __shared__ int len_s[16];

  const int tid  = threadIdx.x;
  const int lane = tid & 31;
  const int wave = tid >> 5;
  const int g    = lane >> 4, nloc = lane & 15;

  const int dir = blockIdx.x;
  const size_t QSZ = 768 * 768, WSZ = 768 * 2304;
  const _Float16* Qp = wbase + (size_t)dir * (2 * QSZ + 2 * WSZ);
  const _Float16* Rp = Qp + QSZ;
  const _Float16* Wp = Rp + QSZ;   // Wih^T packed (768 x 2304)
  const _Float16* Up = Wp + WSZ;   // Whh^T packed (768 x 2304)
  const float* bih = dir ? bih_bw : bih_fw;
  const float* bhh = dir ? bhh_bw : bhh_fw;
  float* dst = dir ? out_bw : out_fw;

  if (tid < 16) len_s[tid] = lengths[tid];
  for (int i = tid; i < 16 * 768; i += NTHREADS) h_s[i] = 0.f;

  for (int step = 0; step < TT; ++step) {
    const int t = dir ? (TT - 1 - step) : step;

    // Stage x_t into LDS (coalesced). mog_gemm's leading barrier covers it.
    for (int i = tid; i < 16 * 768; i += NTHREADS) {
      int m = i / 768, n = i - m * 768;
      xt_s[i] = xg[((size_t)m * TT + t) * DD + n];
    }

    // Mogrifier (2 rounds):
    mog_gemm(h_s,  Qp, xt_s, xm_s, wave, lane);  // xm = 2s(h @Q)*xt
    mog_gemm(xm_s, Rp, h_s,  hm_s, wave, lane);  // hm = 2s(xm@R)*h
    mog_gemm(hm_s, Qp, xm_s, xm_s, wave, lane);  // xm = 2s(hm@Q)*xm
    mog_gemm(xm_s, Rp, hm_s, hm_s, wave, lane);  // hm = 2s(xm@R)*hm

    // gh = hm @ Whh^T + bhh  -> gh_s (f16).  144 N-tiles, 9 per wave.
    __syncthreads();
    {
      v8f acc[9] = {};
      for (int kt = 0; kt < KT; ++kt) {
        v16h a = load_a(hm_s, kt * 32, lane);
#pragma unroll
        for (int i = 0; i < 9; ++i) {
          v16h b = load_b(Up, (wave * 9 + i) * KT + kt, lane);
          acc[i] = wmma16(a, b, acc[i]);
        }
      }
#pragma unroll
      for (int i = 0; i < 9; ++i) {
        const int n = (wave * 9 + i) * 16 + nloc;
        float bb = bhh[n];
#pragma unroll
        for (int v = 0; v < 8; ++v)
          gh_s[(v + 8 * g) * 2304 + n] = (_Float16)(acc[i][v] + bb);
      }
    }

    // gi (in registers, r/z/n chunks per wave-column-group) + fused GRU gates.
    __syncthreads();
    {
      v8f aR[3] = {}, aZ[3] = {}, aN[3] = {};
      for (int kt = 0; kt < KT; ++kt) {
        v16h a = load_a(xm_s, kt * 32, lane);
#pragma unroll
        for (int i = 0; i < 3; ++i) {
          const int ntb = wave * 3 + i;
          aR[i] = wmma16(a, load_b(Wp, (ntb)      * KT + kt, lane), aR[i]);
          aZ[i] = wmma16(a, load_b(Wp, (48 + ntb) * KT + kt, lane), aZ[i]);
          aN[i] = wmma16(a, load_b(Wp, (96 + ntb) * KT + kt, lane), aN[i]);
        }
      }
#pragma unroll
      for (int i = 0; i < 3; ++i) {
        const int n = (wave * 3 + i) * 16 + nloc;
        float bir = bih[n], biz = bih[768 + n], bin = bih[1536 + n];
#pragma unroll
        for (int v = 0; v < 8; ++v) {
          const int m = v + 8 * g;
          float ir  = aR[i][v] + bir;
          float iz  = aZ[i][v] + biz;
          float inn = aN[i][v] + bin;
          float hr = (float)gh_s[m * 2304 + n];
          float hz = (float)gh_s[m * 2304 + 768 + n];
          float hn = (float)gh_s[m * 2304 + 1536 + n];
          float r  = sigf(ir + hr);
          float z  = sigf(iz + hz);
          float nn = tanh_fast(inn + r * hn);
          float hmv  = hm_s[m * 768 + n];
          float hnew = (1.f - z) * nn + z * hmv;
          bool valid = t < len_s[m];
          float hold = h_s[m * 768 + n];
          h_s[m * 768 + n] = valid ? hnew : hold;   // element owned by this lane
          dst[((size_t)m * TT + t) * HH + n] = valid ? hnew : 0.f;
        }
      }
    }
    __syncthreads();
  }
}

// Pack weight matrix (logical B-matrix element B[k][n]) into per-lane WMMA
// fragment order f16: dst[(nt*KT_ + kt)*512 + lane*16 + j] = B[kt*32 + (lane>>4)*16 + j][nt*16 + (lane&15)]
__global__ void pack_w_kernel(const float* __restrict__ src,
                              _Float16* __restrict__ dst,
                              int K, int N, int ld, int transpose) {
  int tid = blockIdx.x * blockDim.x + threadIdx.x;
  int total = K * N;
  if (tid >= total) return;
  int frag = tid >> 9;
  int r    = tid & 511;
  int lane = r >> 4, j = r & 15;
  int KT_  = K >> 5;
  int kt = frag % KT_, nt = frag / KT_;
  int k = kt * 32 + (lane >> 4) * 16 + j;
  int n = nt * 16 + (lane & 15);
  float v = transpose ? src[(size_t)n * ld + k] : src[(size_t)k * ld + n];
  dst[tid] = (_Float16)v;
}

__global__ void add_out_kernel(float* __restrict__ out,
                               const float* __restrict__ add, int n) {
  int i = blockIdx.x * blockDim.x + threadIdx.x;
  if (i < n) out[i] += add[i];
}

extern "C" void kernel_launch(void* const* d_in, const int* in_sizes, int n_in,
                              void* d_out, int out_size, void* d_ws, size_t ws_size,
                              hipStream_t stream) {
  const float* x       = (const float*)d_in[0];
  const int*   lengths = (const int*)  d_in[1];
  const float* Q       = (const float*)d_in[2];
  const float* R       = (const float*)d_in[3];
  const float* Wih_fw  = (const float*)d_in[4];
  const float* Whh_fw  = (const float*)d_in[5];
  const float* bih_fw  = (const float*)d_in[6];
  const float* bhh_fw  = (const float*)d_in[7];
  const float* Wih_bw  = (const float*)d_in[8];
  const float* Whh_bw  = (const float*)d_in[9];
  const float* bih_bw  = (const float*)d_in[10];
  const float* bhh_bw  = (const float*)d_in[11];

  const size_t QSZ = 768 * 768, WSZ = 768 * 2304;
  const size_t PER_DIR = 2 * QSZ + 2 * WSZ;       // f16 elements per direction
  _Float16* wp = (_Float16*)d_ws;
  _Float16* Qp0 = wp;             _Float16* Qp1 = wp + PER_DIR;
  _Float16* Rp0 = Qp0 + QSZ;      _Float16* Rp1 = Qp1 + QSZ;
  _Float16* Wp0 = Rp0 + QSZ;      _Float16* Wp1 = Rp1 + QSZ;
  _Float16* Up0 = Wp0 + WSZ;      _Float16* Up1 = Wp1 + WSZ;
  float* bwbuf = (float*)((char*)d_ws + 2 * PER_DIR * sizeof(_Float16));

  const int PT = 256;
  dim3 gq((unsigned)((QSZ + PT - 1) / PT)), gw((unsigned)((WSZ + PT - 1) / PT));
  // h@Q uses Q[dir] (H x D) directly; xm@R uses R[dir] (D x H) directly.
  pack_w_kernel<<<gq, PT, 0, stream>>>(Q,             Qp0, 768, 768,  768, 0);
  pack_w_kernel<<<gq, PT, 0, stream>>>(Q + QSZ,       Qp1, 768, 768,  768, 0);
  pack_w_kernel<<<gq, PT, 0, stream>>>(R,             Rp0, 768, 768,  768, 0);
  pack_w_kernel<<<gq, PT, 0, stream>>>(R + QSZ,       Rp1, 768, 768,  768, 0);
  // Wih/Whh are (2304 x 768); GEMM uses their transpose (768 x 2304).
  pack_w_kernel<<<gw, PT, 0, stream>>>(Wih_fw, Wp0, 768, 2304, 768, 1);
  pack_w_kernel<<<gw, PT, 0, stream>>>(Whh_fw, Up0, 768, 2304, 768, 1);
  pack_w_kernel<<<gw, PT, 0, stream>>>(Wih_bw, Wp1, 768, 2304, 768, 1);
  pack_w_kernel<<<gw, PT, 0, stream>>>(Whh_bw, Up1, 768, 2304, 768, 1);

  const size_t shmem = 4 * (16 * 768) * sizeof(float)      // h, xm, hm, xt
                     + (16 * 2304) * sizeof(_Float16);     // gh   => 264KB
  mog_gru_kernel<<<dim3(2), dim3(NTHREADS), shmem, stream>>>(
      x, lengths, wp, bih_fw, bhh_fw, bih_bw, bhh_bw, (float*)d_out, bwbuf);

  const int n = 16 * 1024 * 768;
  add_out_kernel<<<(n + PT - 1) / PT, PT, 0, stream>>>((float*)d_out, bwbuf, n);
}
